// QIS_GAN_x16_64836826301070
// MI455X (gfx1250) — matrix operand order repaired and checked by reading, compiled, not run on hardware
//
#include <hip/hip_runtime.h>
#include <hip/hip_bf16.h>
#include <math.h>

typedef __attribute__((ext_vector_type(16))) _Float16 v16h;
typedef __attribute__((ext_vector_type(8)))  float    v8f;
typedef __attribute__((ext_vector_type(8)))  int      v8i;

union Frag { v16h h; v8i i; };

// 16-bit A-matrix 16x32 layout (CDNA5 ISA): lane m = lane%16,
// VGPR v holds K pair starting at 2*(v&3) + 16*(v>>2) + 8*(lane>=16)
__device__ __forceinline__ int a_koff(int lane, int v) {
    return ((v & 3) << 1) + ((v & 4) ? 16 : 0) + ((lane & 16) ? 8 : 0);
}

// ---------------------------------------------------------------------------
// Weight packer: W (K x N, row-major f32) -> f16 B-fragment stream.
// Dense 16-bit B 32x16 per tile: lane n = lane%16, VGPR v holds K pair at
// 2*(v&3) + 16*(v>>2) + 8*(lane>=16) (same K order as the A fragment).
// Output dword index: ((nt*nKt + kt)*32 + lane)*8 + v
// ---------------------------------------------------------------------------
__global__ void pack_b_kernel(const float* __restrict__ W, int K, int N,
                              int nKt, int nNt, unsigned* __restrict__ out)
{
    int total = nNt * nKt * 256;
    for (int o = blockIdx.x * blockDim.x + threadIdx.x; o < total;
         o += gridDim.x * blockDim.x) {
        int v    = o & 7;
        int lane = (o >> 3) & 31;
        int kt   = (o >> 8) % nKt;
        int nt   = o / (nKt << 8);
        int n  = (nt << 4) + (lane & 15);
        int k0 = (kt << 5) + ((v & 3) << 1) + ((v & 4) ? 16 : 0) + ((lane & 16) ? 8 : 0);
        _Float16 lo = (k0     < K && n < N) ? (_Float16)W[k0 * N + n]       : (_Float16)0.f;
        _Float16 hi = (k0 + 1 < K && n < N) ? (_Float16)W[(k0 + 1) * N + n] : (_Float16)0.f;
        unsigned short ulo, uhi;
        __builtin_memcpy(&ulo, &lo, 2);
        __builtin_memcpy(&uhi, &hi, 2);
        out[o] = (unsigned)ulo | ((unsigned)uhi << 16);
    }
}

// ---------------------------------------------------------------------------
// Half-pixel bilinear upsample (jax.image.resize 'bilinear', edge-clamped)
// ---------------------------------------------------------------------------
__global__ void bilinear_up_kernel(const float* __restrict__ in, float* __restrict__ out,
                                   int B, int C, int HI, int WI, int HO, int WO)
{
    int total = B * C * HO * WO;
    for (int i = blockIdx.x * blockDim.x + threadIdx.x; i < total;
         i += gridDim.x * blockDim.x) {
        int x = i % WO; int t = i / WO;
        int y = t % HO; t /= HO;
        int c = t % C;  int b = t / C;
        float sy = (y + 0.5f) * (float)HI / (float)HO - 0.5f;
        float sx = (x + 0.5f) * (float)WI / (float)WO - 0.5f;
        float fy = floorf(sy), fx = floorf(sx);
        int y0 = (int)fy, x0 = (int)fx;
        float wy = sy - fy, wx = sx - fx;
        int y0c = min(max(y0, 0), HI - 1), y1c = min(max(y0 + 1, 0), HI - 1);
        int x0c = min(max(x0, 0), WI - 1), x1c = min(max(x0 + 1, 0), WI - 1);
        const float* src = in + ((size_t)(b * C + c) * HI) * WI;
        float v00 = src[y0c * WI + x0c], v01 = src[y0c * WI + x1c];
        float v10 = src[y1c * WI + x0c], v11 = src[y1c * WI + x1c];
        float v0 = v00 + (v01 - v00) * wx;
        float v1 = v10 + (v11 - v10) * wx;
        out[i] = v0 + (v1 - v0) * wy;
    }
}

// ---------------------------------------------------------------------------
// 3x3 SAME conv + ReLU; input channels concatenated from inA (cA ch) and inB (cB ch)
// ---------------------------------------------------------------------------
__global__ void conv3x3_relu_kernel(const float* __restrict__ inA, int cA,
                                    const float* __restrict__ inB, int cB,
                                    const float* __restrict__ Wt, const float* __restrict__ bias,
                                    float* __restrict__ out, int B, int H, int Wd, int OC)
{
    int IC = cA + cB;
    int total = B * OC * H * Wd;
    for (int i = blockIdx.x * blockDim.x + threadIdx.x; i < total;
         i += gridDim.x * blockDim.x) {
        int x = i % Wd; int t = i / Wd;
        int y = t % H;  t /= H;
        int oc = t % OC; int b = t / OC;
        float acc = bias[oc];
        for (int ic = 0; ic < IC; ++ic) {
            const float* src = (ic < cA)
                ? (inA + (size_t)(b * cA + ic) * H * Wd)
                : (inB + (size_t)(b * cB + (ic - cA)) * H * Wd);
            const float* wk = Wt + (size_t)(oc * IC + ic) * 9;
            #pragma unroll
            for (int ky = 0; ky < 3; ++ky) {
                int yy = y + ky - 1;
                if (yy < 0 || yy >= H) continue;
                #pragma unroll
                for (int kx = 0; kx < 3; ++kx) {
                    int xx = x + kx - 1;
                    if (xx < 0 || xx >= Wd) continue;
                    acc += src[yy * Wd + xx] * wk[ky * 3 + kx];
                }
            }
        }
        out[i] = fmaxf(acc, 0.f);
    }
}

// ---------------------------------------------------------------------------
// kxk max pool, stride k
// ---------------------------------------------------------------------------
__global__ void maxpool_kernel(const float* __restrict__ in, float* __restrict__ out,
                               int BC, int H, int W, int k)
{
    int Ho = H / k, Wo = W / k;
    int total = BC * Ho * Wo;
    for (int i = blockIdx.x * blockDim.x + threadIdx.x; i < total;
         i += gridDim.x * blockDim.x) {
        int x = i % Wo; int t = i / Wo;
        int y = t % Ho; int bc = t / Ho;
        const float* src = in + ((size_t)bc * H + (size_t)y * k) * W + (size_t)x * k;
        float mx = -INFINITY;
        for (int ky = 0; ky < k; ++ky)
            for (int kx = 0; kx < k; ++kx)
                mx = fmaxf(mx, src[ky * W + kx]);
        out[i] = mx;
    }
}

__global__ void add_kernel(const float* __restrict__ a, const float* __restrict__ b,
                           float* __restrict__ out, int total)
{
    for (int i = blockIdx.x * blockDim.x + threadIdx.x; i < total;
         i += gridDim.x * blockDim.x)
        out[i] = a[i] + b[i];
}

// ---------------------------------------------------------------------------
// Fused LIIF-style query stage. 128 threads = 4 waves; wave b = branch b
// (vx = b&2 ? +1 : -1, vy = b&1 ? +1 : -1). Each block handles 16 query
// positions of one batch image. MLP layers run as 16x16x32 f16 WMMA.
//   DINP = padded input dim (160 for stage16, 128 for the others)
// ---------------------------------------------------------------------------
template <int DINP>
__global__ __launch_bounds__(128)
void query_kernel(const float* __restrict__ feat, int c, int h, int w,
                  const float* __restrict__ guide, int cg, int Hg, int Wg,
                  const unsigned* __restrict__ W1p, const float* __restrict__ b1,
                  const unsigned* __restrict__ W2p, const float* __restrict__ b2,
                  const unsigned* __restrict__ W3p, const float* __restrict__ b3,
                  int out_ch, int out_pad, float* __restrict__ out)
{
    __shared__ __align__(16) _Float16 sX[4][16 * DINP];   // X, later reused as H2
    __shared__ __align__(16) _Float16 sH1[4][16 * 256];   // H1, later reused as P(f32)

    const int lane = threadIdx.x & 31;
    const int wave = threadIdx.x >> 5;
    const int ntiles = (Hg * Wg) >> 4;
    const int batch = blockIdx.x / ntiles;
    const int tile  = blockIdx.x - batch * ntiles;

    const float vxo = (wave & 2) ? 1.f : -1.f;
    const float vyo = (wave & 1) ? 1.f : -1.f;
    const float invh = 1.f / (float)h, invw = 1.f / (float)w;

    // ---- gather per-row input vector [q_feat | q_guide | rel | pad] as f16 ----
    for (int idx = lane; idx < 16 * DINP; idx += 32) {
        int m = idx / DINP, k = idx - m * DINP;
        int p = (tile << 4) + m;
        int py = p / Wg, px = p - py * Wg;
        float c0 = -1.f + (2.f * py + 1.f) / (float)Hg;
        float c1 = -1.f + (2.f * px + 1.f) / (float)Wg;
        float fy = ((c0 + vxo * invh + 1.f) * (float)h - 1.f) * 0.5f;
        float fx = ((c1 + vyo * invw + 1.f) * (float)w - 1.f) * 0.5f;
        int iy = (int)rintf(fy), ix = (int)rintf(fx);
        bool valid = (iy >= 0) && (iy < h) && (ix >= 0) && (ix < w);
        int iyc = min(max(iy, 0), h - 1);
        int ixc = min(max(ix, 0), w - 1);
        float val;
        if (k < c) {
            val = valid ? feat[((size_t)(batch * c + k) * h + iyc) * w + ixc] : 0.f;
        } else if (k < c + cg) {
            val = guide[((size_t)(batch * cg + (k - c)) * Hg + py) * Wg + px];
        } else if (k == c + cg) {
            float qc = valid ? (-1.f + (2.f * iyc + 1.f) * invh) : 0.f;
            val = (c0 - qc) * (float)h;
        } else if (k == c + cg + 1) {
            float qc = valid ? (-1.f + (2.f * ixc + 1.f) * invw) : 0.f;
            val = (c1 - qc) * (float)w;
        } else {
            val = 0.f;
        }
        sX[wave][idx] = (_Float16)val;
    }
    __syncthreads();

    const int m    = lane & 15;           // A row
    const int nsel = lane & 15;           // C column
    const int mhi  = (lane & 16) ? 8 : 0; // C row offset

    // ---- layer 1: X(16 x DINP) @ W1(DINP x 256) + b1, ReLU -> H1 ----
    constexpr int nK1 = DINP / 32;
    for (int nt = 0; nt < 16; ++nt) {
        v8f acc = {};
        #pragma unroll
        for (int kt = 0; kt < nK1; ++kt) {
            Frag A, B;
            const _Float16* arow = &sX[wave][m * DINP + kt * 32];
            #pragma unroll
            for (int v = 0; v < 8; ++v)
                A.i[v] = *(const int*)(arow + a_koff(lane, v));
            B.i = *(const v8i*)(W1p + (((nt * nK1 + kt) * 32 + lane) << 3));
            acc = __builtin_amdgcn_wmma_f32_16x16x32_f16(false, A.h, false, B.h,
                                                         (short)0, acc, false, false);
        }
        int n = (nt << 4) + nsel;
        float bias = b1[n];
        #pragma unroll
        for (int v = 0; v < 8; ++v)
            sH1[wave][(v + mhi) * 256 + n] = (_Float16)fmaxf(acc[v] + bias, 0.f);
    }
    __syncthreads();

    // ---- layer 2: H1(16 x 256) @ W2(256 x 128) + b2, ReLU -> H2 (in sX region) ----
    _Float16* H2 = &sX[wave][0];
    for (int nt = 0; nt < 8; ++nt) {
        v8f acc = {};
        #pragma unroll
        for (int kt = 0; kt < 8; ++kt) {
            Frag A, B;
            const _Float16* arow = &sH1[wave][m * 256 + kt * 32];
            #pragma unroll
            for (int v = 0; v < 8; ++v)
                A.i[v] = *(const int*)(arow + a_koff(lane, v));
            B.i = *(const v8i*)(W2p + (((nt * 8 + kt) * 32 + lane) << 3));
            acc = __builtin_amdgcn_wmma_f32_16x16x32_f16(false, A.h, false, B.h,
                                                         (short)0, acc, false, false);
        }
        int n = (nt << 4) + nsel;
        float bias = b2[n];
        #pragma unroll
        for (int v = 0; v < 8; ++v)
            H2[(v + mhi) * 128 + n] = (_Float16)fmaxf(acc[v] + bias, 0.f);
    }
    __syncthreads();

    // ---- layer 3: H2(16 x 128) @ W3(128 x out_pad) + b3 -> P f32 (in sH1 region) ----
    float* P = (float*)&sH1[wave][0];
    int nN3 = out_pad >> 4;
    for (int nt = 0; nt < nN3; ++nt) {
        v8f acc = {};
        #pragma unroll
        for (int kt = 0; kt < 4; ++kt) {
            Frag A, B;
            const _Float16* arow = H2 + m * 128 + kt * 32;
            #pragma unroll
            for (int v = 0; v < 8; ++v)
                A.i[v] = *(const int*)(arow + a_koff(lane, v));
            B.i = *(const v8i*)(W3p + (((nt * 4 + kt) * 32 + lane) << 3));
            acc = __builtin_amdgcn_wmma_f32_16x16x32_f16(false, A.h, false, B.h,
                                                         (short)0, acc, false, false);
        }
        int n = (nt << 4) + nsel;
        float bias = (n < out_ch) ? b3[n] : 0.f;
        #pragma unroll
        for (int v = 0; v < 8; ++v)
            P[(v + mhi) * out_pad + n] = acc[v] + bias;
    }
    __syncthreads();

    // ---- softmax over 4 branches on the last channel; weighted sum of the rest ----
    int outc = out_ch - 1;
    for (int idx = threadIdx.x; idx < 16 * outc; idx += 128) {
        int mpos = idx / outc, ch = idx - mpos * outc;
        const float* P0 = (const float*)&sH1[0][0];
        const float* P1 = (const float*)&sH1[1][0];
        const float* P2 = (const float*)&sH1[2][0];
        const float* P3 = (const float*)&sH1[3][0];
        int lrow = mpos * out_pad;
        float l0 = P0[lrow + outc], l1 = P1[lrow + outc];
        float l2 = P2[lrow + outc], l3 = P3[lrow + outc];
        float mx = fmaxf(fmaxf(l0, l1), fmaxf(l2, l3));
        float e0 = __expf(l0 - mx), e1 = __expf(l1 - mx);
        float e2 = __expf(l2 - mx), e3 = __expf(l3 - mx);
        float s = e0 + e1 + e2 + e3;
        float val = (P0[lrow + ch] * e0 + P1[lrow + ch] * e1 +
                     P2[lrow + ch] * e2 + P3[lrow + ch] * e3) / s;
        int p = (tile << 4) + mpos;
        int py = p / Wg, px = p - py * Wg;
        out[((size_t)(batch * outc + ch) * Hg + py) * Wg + px] = val;
    }
}

// ---------------------------------------------------------------------------
// host launcher
// ---------------------------------------------------------------------------
static inline int gblocks(int total) { return (total + 255) / 256; }

extern "C" void kernel_launch(void* const* d_in, const int* in_sizes, int n_in,
                              void* d_out, int out_size, void* d_ws, size_t ws_size,
                              hipStream_t stream)
{
    (void)in_sizes; (void)n_in; (void)out_size; (void)ws_size;

    const float* IN[30];
    for (int i = 0; i < 30; ++i) IN[i] = (const float*)d_in[i];
    const float* LR    = IN[0];   // (2,31,16,16)
    const float* MS    = IN[1];   // (2,3,256,256)
    const float* spa_w = IN[2];
    const float* spa_b = IN[3];
    const float* spe_w = IN[4];
    const float* spe_b = IN[5];
    // imnet16: 6..11  imnet32: 12..17  imnet64: 18..23  imnet128: 24..29
    // order per mlp: w1,b1,w2,b2,w3,b3

    char* base = (char*)d_ws;
    size_t off = 0;
    auto carve = [&](size_t elems, size_t esize) -> void* {
        void* p = base + off;
        off += ((elems * esize + 255) & ~(size_t)255);
        return p;
    };

    float* lms    = (float*)carve(2u * 31 * 256 * 256, 4);
    float* hr_spa = (float*)carve(2u * 64 * 256 * 256, 4);
    float* g16    = (float*)carve(2u * 64 * 32 * 32, 4);
    float* g32    = (float*)carve(2u * 64 * 64 * 64, 4);
    float* g64    = (float*)carve(2u * 64 * 128 * 128, 4);
    float* lr_spe = (float*)carve(2u * 64 * 16 * 16, 4);
    float* f16b   = (float*)carve(2u * 32 * 32 * 32, 4);
    float* f32b   = (float*)carve(2u * 32 * 64 * 64, 4);
    float* f64b   = (float*)carve(2u * 32 * 128 * 128, 4);
    float* f128b  = (float*)carve(2u * 31 * 256 * 256, 4);
    unsigned* w1p16  = (unsigned*)carve(16 * 5 * 256, 4);
    unsigned* w2p16  = (unsigned*)carve(8 * 8 * 256, 4);
    unsigned* w3p16  = (unsigned*)carve(3 * 4 * 256, 4);
    unsigned* w1p32  = (unsigned*)carve(16 * 4 * 256, 4);
    unsigned* w2p32  = (unsigned*)carve(8 * 8 * 256, 4);
    unsigned* w3p32  = (unsigned*)carve(3 * 4 * 256, 4);
    unsigned* w1p64  = (unsigned*)carve(16 * 4 * 256, 4);
    unsigned* w2p64  = (unsigned*)carve(8 * 8 * 256, 4);
    unsigned* w3p64  = (unsigned*)carve(3 * 4 * 256, 4);
    unsigned* w1p128 = (unsigned*)carve(16 * 4 * 256, 4);
    unsigned* w2p128 = (unsigned*)carve(8 * 8 * 256, 4);
    unsigned* w3p128 = (unsigned*)carve(2 * 4 * 256, 4);

    // --- pack MLP weights into WMMA B-fragment f16 layout ---
    pack_b_kernel<<<gblocks(16*5*256), 256, 0, stream>>>(IN[6],  130, 256, 5, 16, w1p16);
    pack_b_kernel<<<gblocks(8*8*256),  256, 0, stream>>>(IN[8],  256, 128, 8,  8, w2p16);
    pack_b_kernel<<<gblocks(3*4*256),  256, 0, stream>>>(IN[10], 128,  33, 4,  3, w3p16);
    pack_b_kernel<<<gblocks(16*4*256), 256, 0, stream>>>(IN[12],  98, 256, 4, 16, w1p32);
    pack_b_kernel<<<gblocks(8*8*256),  256, 0, stream>>>(IN[14], 256, 128, 8,  8, w2p32);
    pack_b_kernel<<<gblocks(3*4*256),  256, 0, stream>>>(IN[16], 128,  33, 4,  3, w3p32);
    pack_b_kernel<<<gblocks(16*4*256), 256, 0, stream>>>(IN[18],  98, 256, 4, 16, w1p64);
    pack_b_kernel<<<gblocks(8*8*256),  256, 0, stream>>>(IN[20], 256, 128, 8,  8, w2p64);
    pack_b_kernel<<<gblocks(3*4*256),  256, 0, stream>>>(IN[22], 128,  33, 4,  3, w3p64);
    pack_b_kernel<<<gblocks(16*4*256), 256, 0, stream>>>(IN[24],  98, 256, 4, 16, w1p128);
    pack_b_kernel<<<gblocks(8*8*256),  256, 0, stream>>>(IN[26], 256, 128, 8,  8, w2p128);
    pack_b_kernel<<<gblocks(2*4*256),  256, 0, stream>>>(IN[28], 128,  32, 4,  2, w3p128);

    // --- lms = bilinear x16 upsample of LR_HSI ---
    bilinear_up_kernel<<<gblocks(2*31*256*256), 256, 0, stream>>>(
        LR, lms, 2, 31, 16, 16, 256, 256);

    // --- hr_spa = relu(conv3x3(concat[HR_MSI, lms])) ---
    conv3x3_relu_kernel<<<gblocks(2*64*256*256), 256, 0, stream>>>(
        MS, 3, lms, 31, spa_w, spa_b, hr_spa, 2, 256, 256, 64);

    // --- guides ---
    maxpool_kernel<<<gblocks(128*32*32),   256, 0, stream>>>(hr_spa, g16, 128, 256, 256, 8);
    maxpool_kernel<<<gblocks(128*64*64),   256, 0, stream>>>(hr_spa, g32, 128, 256, 256, 4);
    maxpool_kernel<<<gblocks(128*128*128), 256, 0, stream>>>(hr_spa, g64, 128, 256, 256, 2);

    // --- lr_spe = relu(conv3x3(LR_HSI)) ---
    conv3x3_relu_kernel<<<gblocks(2*64*16*16), 256, 0, stream>>>(
        LR, 31, (const float*)nullptr, 0, spe_w, spe_b, lr_spe, 2, 16, 16, 64);

    // --- query stages (fused gather + 3-layer WMMA MLP + softmax combine) ---
    query_kernel<160><<<2 * (32 * 32 / 16), 128, 0, stream>>>(
        lr_spe, 64, 16, 16, g16, 64, 32, 32,
        w1p16, IN[7], w2p16, IN[9], w3p16, IN[11], 33, 48, f16b);
    query_kernel<128><<<2 * (64 * 64 / 16), 128, 0, stream>>>(
        f16b, 32, 32, 32, g32, 64, 64, 64,
        w1p32, IN[13], w2p32, IN[15], w3p32, IN[17], 33, 48, f32b);
    query_kernel<128><<<2 * (128 * 128 / 16), 128, 0, stream>>>(
        f32b, 32, 64, 64, g64, 64, 128, 128,
        w1p64, IN[19], w2p64, IN[21], w3p64, IN[23], 33, 48, f64b);
    query_kernel<128><<<2 * (256 * 256 / 16), 128, 0, stream>>>(
        f64b, 32, 128, 128, hr_spa, 64, 256, 256,
        w1p128, IN[25], w2p128, IN[27], w3p128, IN[29], 32, 32, f128b);

    // --- final: out = lms + f128 ---
    add_kernel<<<gblocks(2*31*256*256), 256, 0, stream>>>(
        lms, f128b, (float*)d_out, 2 * 31 * 256 * 256);
}
